// ReadingLayer_17377437679860
// MI455X (gfx1250) — matrix operand order repaired and checked by reading, not compiled, not run on hardware
//
#include <hip/hip_runtime.h>
#include <stdint.h>

#define N_SLOTS 65536
#define D_FEAT  64
#define BATCH   2048

#define TB      16                  // batch rows per workgroup
#define TK      128                 // K (slots) per staged iteration
#define NCHUNK  16                  // N-dimension split for parallelism
#define CHUNK   (N_SLOTS / NCHUNK)  // 4096 slots per chunk
#define APITCH  136                 // padded LDS pitch (bf16 units), A tile
#define MPITCH  136                 // padded LDS pitch (bf16 units), B tile

typedef __attribute__((ext_vector_type(16))) __bf16 v16bf;
typedef __attribute__((ext_vector_type(8)))  float  v8f;

__device__ __forceinline__ uint32_t pack_bf16(float a, float b) {
  uint32_t ua = __float_as_uint(a);
  uint32_t ub = __float_as_uint(b);
  ua = (ua + 0x7FFFu + ((ua >> 16) & 1u)) >> 16;   // round-to-nearest-even
  ub = (ub + 0x7FFFu + ((ub >> 16) & 1u)) >> 16;
  return ua | (ub << 16);
}

// ---------------------------------------------------------------------------
// Prepass: memory [65536,64] f32  ->  Mtg [64][65536] bf16 (transposed copy).
// 24 MB of traffic total; makes the hot kernel's B tile a pure async DMA.
// ---------------------------------------------------------------------------
__global__ void __launch_bounds__(256)
transpose_bf16_kernel(const float* __restrict__ memory, uint16_t* __restrict__ Mtg) {
  const int idx = blockIdx.x * blockDim.x + threadIdx.x;  // 64 * 4096 threads
  const int f  = idx >> 12;          // feature 0..63
  const int s0 = (idx & 4095) * 16;  // 16 consecutive slots
  uint32_t w[8];
  #pragma unroll
  for (int j = 0; j < 8; ++j) {
    float a = memory[(size_t)(s0 + 2 * j)     * D_FEAT + f];
    float b = memory[(size_t)(s0 + 2 * j + 1) * D_FEAT + f];
    w[j] = pack_bf16(a, b);
  }
  uint16_t* dst = Mtg + (size_t)f * N_SLOTS + s0;
  ((uint4*)dst)[0] = make_uint4(w[0], w[1], w[2], w[3]);
  ((uint4*)dst)[1] = make_uint4(w[4], w[5], w[6], w[7]);
}

// ---------------------------------------------------------------------------
// Zero numerator accumulator (d_out) and denominator accumulator (ssum).
// ---------------------------------------------------------------------------
__global__ void zero_kernel(float* __restrict__ out, float* __restrict__ ssum) {
  int i = blockIdx.x * blockDim.x + threadIdx.x;
  if (i < BATCH * D_FEAT) out[i] = 0.0f;
  if (i < BATCH)          ssum[i] = 0.0f;
}

// ---------------------------------------------------------------------------
// Single-pass softmax-read. Grid: 128 batch-tiles x NCHUNK chunks,
// 128 threads = 4 waves; wave w owns output features [16w, 16w+16).
// A tile: p = exp(w) -> bf16 (unnormalized; fp32 row-sum kept on the side).
// B tile: PRET ? async DMA from pre-transposed bf16 Mtg (ASYNCcnt path)
//              : in-kernel f32->bf16 transpose staging.
// 4 x v_wmma_f32_16x16x32_bf16 per wave per 128-K iteration.
// ---------------------------------------------------------------------------
template <bool PRET>
__global__ void __launch_bounds__(128)
read_wmma_kernel(const float* __restrict__ memory,
                 const uint16_t* __restrict__ Mtg,
                 const float* __restrict__ weight,
                 float* __restrict__ out,
                 float* __restrict__ ssum) {
  __shared__ __align__(16) uint16_t Ab[TB * APITCH];
  __shared__ __align__(16) uint16_t Mt[D_FEAT * MPITCH];
  __shared__ float sred[128];

  const int t     = threadIdx.x;    // 0..127
  const int lane  = t & 31;
  const int wave  = t >> 5;         // feature column group 0..3
  const int l15   = lane & 15;
  const int lhalf = lane >> 4;

  const int bt = blockIdx.x & 127;  // batch tile
  const int cn = blockIdx.x >> 7;   // chunk index
  const int b0 = bt * TB;
  const int n0 = cn * CHUNK;

  // A-staging mapping: thread t handles 16 fp32 of row ar per iteration.
  const int ar = t >> 3;            // 0..15
  const int ac = (t & 7) * 4;
  const float* __restrict__ wrow = weight + (size_t)(b0 + ar) * N_SLOTS + n0;

  // Async B-staging mapping (PRET): each wave DMAs its own 16 rows.
  const int brow_st = wave * 16 + (lane >> 1);       // feature row staged
  const uint32_t lds_row =
      (uint32_t)(uintptr_t)&Mt[brow_st * MPITCH] + (lane & 1) * 128;
  const uint32_t goff =
      (uint32_t)brow_st * (uint32_t)(N_SLOTS * 2) + (lane & 1) * 128;

  // Fallback B-staging mapping.
  const int f2 = t & 31;
  const int kq = t >> 5;

  v8f acc = {0.f, 0.f, 0.f, 0.f, 0.f, 0.f, 0.f, 0.f};
  float srow = 0.0f;                // partial denominator for row ar

  for (int kb = 0; kb < CHUNK / TK; ++kb) {
    const int k0 = n0 + kb * TK;

    // ---- stage A: exp(w) -> bf16, plus fp32 row-sum ----
    #pragma unroll
    for (int p = 0; p < 4; ++p) {
      const int col = ac + p * 32;
      float4 v = *(const float4*)(wrow + kb * TK + col);
      float p0 = __expf(v.x);
      float p1 = __expf(v.y);
      float p2 = __expf(v.z);
      float p3 = __expf(v.w);
      srow += (p0 + p1) + (p2 + p3);
      *(uint2*)&Ab[ar * APITCH + col] =
          make_uint2(pack_bf16(p0, p1), pack_bf16(p2, p3));
    }

    // ---- stage B ----
    if (PRET) {
      // Async DMA: 16 rows x 256B per wave from Mtg[f][k0..k0+127] (bf16).
      const uint16_t* sbase = Mtg + k0;   // uniform SGPR base
      #pragma unroll
      for (int j = 0; j < 8; ++j) {
        asm volatile("global_load_async_to_lds_b128 %0, %1, %2 offset:%3"
                     :: "v"(lds_row), "v"(goff), "s"(sbase), "n"(j * 16)
                     : "memory");
      }
      asm volatile("s_wait_asynccnt 0x0" ::: "memory");
    } else {
      #pragma unroll 4
      for (int q = 0; q < 16; ++q) {
        const int k = (kq + q * 4) * 2;
        const float* mp = memory + (size_t)(k0 + k) * D_FEAT + f2 * 2;
        float2 a = *(const float2*)mp;
        float2 b = *(const float2*)(mp + D_FEAT);
        *(uint32_t*)&Mt[(2 * f2)     * MPITCH + k] = pack_bf16(a.x, b.x);
        *(uint32_t*)&Mt[(2 * f2 + 1) * MPITCH + k] = pack_bf16(a.y, b.y);
      }
    }

    __syncthreads();

    // ---- 4 WMMA steps of K=32 ----
    #pragma unroll
    for (int s = 0; s < 4; ++s) {
      union { uint4 q[2]; v16bf v; } a, b;
      const int acol = s * 32 + lhalf * 8;
      a.q[0] = *(const uint4*)&Ab[l15 * APITCH + acol];
      a.q[1] = *(const uint4*)&Ab[l15 * APITCH + acol + 16];
      const int brow = wave * 16 + l15;
      const int bcol = s * 32 + lhalf * 8;
      b.q[0] = *(const uint4*)&Mt[brow * MPITCH + bcol];
      b.q[1] = *(const uint4*)&Mt[brow * MPITCH + bcol + 16];
      acc = __builtin_amdgcn_wmma_f32_16x16x32_bf16(
          false, a.v, false, b.v, (short)0, acc, false, false);
    }

    __syncthreads();
  }

  // ---- accumulate numerator partials ----
  #pragma unroll
  for (int i = 0; i < 8; ++i) {
    const int m = lhalf * 8 + i;   // C/D layout: lanes 0-15 M=i, 16-31 M=8+i
    const int n = wave * 16 + l15;
    unsafeAtomicAdd(&out[(size_t)(b0 + m) * D_FEAT + n], acc[i]);
  }

  // ---- accumulate denominator partials (8 threads share each row) ----
  sred[t] = srow;
  __syncthreads();
  if (t < TB) {
    float tot = 0.0f;
    #pragma unroll
    for (int j = 0; j < 8; ++j) tot += sred[t * 8 + j];
    unsafeAtomicAdd(&ssum[b0 + t], tot);
  }
}

// ---------------------------------------------------------------------------
// Final normalization: out[b,d] /= ssum[b].
// ---------------------------------------------------------------------------
__global__ void normalize_kernel(float* __restrict__ out,
                                 const float* __restrict__ ssum) {
  int i = blockIdx.x * blockDim.x + threadIdx.x;
  if (i < BATCH * D_FEAT) out[i] *= (1.0f / ssum[i >> 6]);
}

extern "C" void kernel_launch(void* const* d_in, const int* in_sizes, int n_in,
                              void* d_out, int out_size, void* d_ws, size_t ws_size,
                              hipStream_t stream) {
  const float* memory = (const float*)d_in[0];   // [65536, 64]
  const float* weight = (const float*)d_in[1];   // [2048, 65536]
  float* out  = (float*)d_out;                   // [2048, 64]

  float*    ssum = (float*)d_ws;                             // [2048]
  uint16_t* Mtg  = (uint16_t*)((char*)d_ws + 8192);          // [64][65536] bf16
  const size_t need_pret = 8192 + (size_t)D_FEAT * N_SLOTS * 2;
  const bool pret = ws_size >= need_pret;

  if (pret) {
    transpose_bf16_kernel<<<(D_FEAT * 4096) / 256, 256, 0, stream>>>(memory, Mtg);
  }
  zero_kernel<<<(BATCH * D_FEAT + 255) / 256, 256, 0, stream>>>(out, ssum);
  if (pret) {
    read_wmma_kernel<true><<<128 * NCHUNK, 128, 0, stream>>>(
        memory, Mtg, weight, out, ssum);
  } else {
    read_wmma_kernel<false><<<128 * NCHUNK, 128, 0, stream>>>(
        memory, Mtg, weight, out, ssum);
  }
  normalize_kernel<<<(BATCH * D_FEAT + 255) / 256, 256, 0, stream>>>(out, ssum);
}
